// Linear4bit_29523605193373
// MI455X (gfx1250) — matrix operand order repaired and checked by reading, not compiled, not run on hardware
//
#include <hip/hip_runtime.h>
#include <hip/hip_bf16.h>

// ---------------------------------------------------------------------------
// NF4 Linear: out[M,N] = X[M,K] * dequant(W)[N,K]^T + bias
//   M = 8192, N = 11008, K = 4096, NF4 block = 64 (along K)
//
// Fast path (needs ~157MB workspace):
//   1) dequant W -> f16 once   (memory bound, removes 64x redundant VALU)
//   2) convert X -> f16 once
//   3) f16 WMMA GEMM: double-buffered LDS tiles filled with
//      global_load_async_to_lds_b128 (ASYNCcnt), 16 wmma per K-step per wave.
// Fallback path: fused dequant GEMM (round-1 kernel).
// ---------------------------------------------------------------------------

typedef __attribute__((ext_vector_type(16))) _Float16 v16h;
typedef __attribute__((ext_vector_type(8)))  _Float16 v8h;
typedef __attribute__((ext_vector_type(4)))  _Float16 v4h;
typedef __attribute__((ext_vector_type(8)))  float    v8f;
typedef __attribute__((ext_vector_type(4)))  float    v4f;
typedef __attribute__((ext_vector_type(4)))  int      v4i;

#define OUT_F    11008
#define IN_F     4096
#define M_TOTAL  8192
#define QBLOCK   64

#define LDA      40            // padded LDS row stride in halves (80B, conflict-free)
#define KSTEPS   (IN_F / 32)   // 128

__device__ const float NF4_CODE_D[16] = {
    -1.0f, -0.6961928009986877f, -0.5250730514526367f, -0.39491748809814453f,
    -0.28444138169288635f, -0.18477343022823334f, -0.09105003625154495f, 0.0f,
    0.07958029955625534f, 0.16093020141124725f, 0.24611230194568634f,
    0.33791524171829224f, 0.44070982933044434f, 0.5626170039176941f,
    0.7229568362236023f, 1.0f};

union FragU {
    v16h v;
    v8h  h[2];
};

// ---- gfx1250 async global->LDS copy (ISA 10.x / 15.18, ASYNCcnt) ----------
__device__ __forceinline__ void async_copy_b128(unsigned lds_off, const void* gptr) {
    asm volatile("global_load_async_to_lds_b128 %0, %1, off"
                 :: "v"(lds_off), "v"(gptr)
                 : "memory");
}
__device__ __forceinline__ void wait_async_all() {
    asm volatile("s_wait_asynccnt 0x0" ::: "memory");
}

// ===========================================================================
// Pass 1: NF4 -> f16 weight dequant (memory bound: 180MB in, 90MB out)
// ===========================================================================
__global__ __launch_bounds__(256)
void nf4_dequant_w_kernel(const int*   __restrict__ w_idx,
                          const float* __restrict__ absmax,
                          _Float16*    __restrict__ wh)
{
    __shared__ float codeLds[16];
    if (threadIdx.x < 16) codeLds[threadIdx.x] = NF4_CODE_D[threadIdx.x];
    __syncthreads();

    const size_t base = ((size_t)blockIdx.x * 256 + threadIdx.x) * 8;  // 8 elems/thread
    const v4i a = *(const v4i*)(w_idx + base);
    const v4i b = *(const v4i*)(w_idx + base + 4);
    const float s = absmax[base >> 6];        // 8 contiguous elems share one block

    v8h o;
    o[0] = (_Float16)(codeLds[a.x] * s);
    o[1] = (_Float16)(codeLds[a.y] * s);
    o[2] = (_Float16)(codeLds[a.z] * s);
    o[3] = (_Float16)(codeLds[a.w] * s);
    o[4] = (_Float16)(codeLds[b.x] * s);
    o[5] = (_Float16)(codeLds[b.y] * s);
    o[6] = (_Float16)(codeLds[b.z] * s);
    o[7] = (_Float16)(codeLds[b.w] * s);
    *(v8h*)(wh + base) = o;
}

// ===========================================================================
// Pass 2: X f32 -> f16 (memory bound: 134MB in, 67MB out)
// ===========================================================================
__global__ __launch_bounds__(256)
void x_to_f16_kernel(const float* __restrict__ x, _Float16* __restrict__ xh)
{
    const size_t base = ((size_t)blockIdx.x * 256 + threadIdx.x) * 8;
    const v4f a = *(const v4f*)(x + base);
    const v4f b = *(const v4f*)(x + base + 4);
    v8h o;
    o[0] = (_Float16)a.x;  o[1] = (_Float16)a.y;
    o[2] = (_Float16)a.z;  o[3] = (_Float16)a.w;
    o[4] = (_Float16)b.x;  o[5] = (_Float16)b.y;
    o[6] = (_Float16)b.z;  o[7] = (_Float16)b.w;
    *(v8h*)(xh + base) = o;
}

// ===========================================================================
// Pass 3: f16 WMMA GEMM, 128(M) x 256(N) workgroup tile, BK=32,
// double-buffered LDS via async copies, 64x64 per wave = 16 wmma / K-step.
// ===========================================================================
#define GBM 128
#define GBN 256
#define ASZ (GBM * LDA)   // halves per A buffer
#define BSZ (GBN * LDA)   // halves per B buffer

__global__ __launch_bounds__(256)
void f16_gemm_wmma_kernel(const _Float16* __restrict__ xh,
                          const _Float16* __restrict__ wh,
                          const float*    __restrict__ bias,
                          float*          __restrict__ out)
{
    __shared__ __align__(16) _Float16 Ah[2 * ASZ];   // 20.0 KB
    __shared__ __align__(16) _Float16 Bh[2 * BSZ];   // 40.0 KB

    const int tid    = threadIdx.x;
    const int lane   = tid & 31;
    const int waveId = tid >> 5;
    const int wm     = waveId >> 2;   // 0..1  (64 rows of M each)
    const int wn     = waveId & 3;    // 0..3  (64 cols of N each)
    const int lgrp   = lane >> 4;     // half-wave
    const int l16    = lane & 15;

    const int mTile = blockIdx.y * GBM;   // 64 tiles
    const int nTile = blockIdx.x * GBN;   // 43 tiles

    const _Float16* aBase = xh + (size_t)mTile * IN_F;
    const _Float16* bBase = wh + (size_t)nTile * IN_F;

    // issue async copies for one 32-wide K slice into LDS buffer `buf`
    auto issue_tile = [&](int ks, int buf) {
        const _Float16* ap = aBase + ks * 32;
        const _Float16* bp = bBase + ks * 32;
#pragma unroll
        for (int r = 0; r < 2; ++r) {               // A: 128 rows x 64B = 512 b128
            const int f   = r * 256 + tid;
            const int row = f >> 2;
            const int c   = f & 3;
            async_copy_b128(
                (unsigned)(size_t)&Ah[buf * ASZ + row * LDA + c * 8],
                ap + (size_t)row * IN_F + c * 8);
        }
#pragma unroll
        for (int r = 0; r < 4; ++r) {               // B: 256 rows x 64B = 1024 b128
            const int f   = r * 256 + tid;
            const int row = f >> 2;
            const int c   = f & 3;
            async_copy_b128(
                (unsigned)(size_t)&Bh[buf * BSZ + row * LDA + c * 8],
                bp + (size_t)row * IN_F + c * 8);
        }
    };

    v8f acc[4][4];
#pragma unroll
    for (int i = 0; i < 4; ++i)
#pragma unroll
        for (int j = 0; j < 4; ++j)
            acc[i][j] = 0.f;

    issue_tile(0, 0);

    for (int ks = 0; ks < KSTEPS; ++ks) {
        const int cur = ks & 1;

        wait_async_all();      // this wave's copies for tile ks landed in LDS
        __syncthreads();       // all waves' copies landed; prev buffer free

        if (ks + 1 < KSTEPS) issue_tile(ks + 1, cur ^ 1);   // overlap with WMMA

        const _Float16* Ab = &Ah[cur * ASZ];
        const _Float16* Bb = &Bh[cur * BSZ];

        // A fragments (16x32 f16): lane = M%16, chunks at K = lgrp*8, 16+lgrp*8
        v16h afrag[4];
#pragma unroll
        for (int i = 0; i < 4; ++i) {
            const int mrow = wm * 64 + i * 16 + l16;
            FragU ua;
            ua.h[0] = *(const v8h*)&Ab[mrow * LDA + lgrp * 8];
            ua.h[1] = *(const v8h*)&Ab[mrow * LDA + 16 + lgrp * 8];
            afrag[i] = ua.v;
        }
        // B fragments (32x16 f16): lane = N%16, chunks at K = lgrp*8, 16+lgrp*8
        v16h bfrag[4];
#pragma unroll
        for (int j = 0; j < 4; ++j) {
            const int nrow = wn * 64 + j * 16 + l16;
            FragU ub;
            ub.h[0] = *(const v8h*)&Bb[nrow * LDA + lgrp * 8];
            ub.h[1] = *(const v8h*)&Bb[nrow * LDA + 16 + lgrp * 8];
            bfrag[j] = ub.v;
        }

#pragma unroll
        for (int i = 0; i < 4; ++i)
#pragma unroll
            for (int j = 0; j < 4; ++j)
                acc[i][j] = __builtin_amdgcn_wmma_f32_16x16x32_f16(
                    false, afrag[i], false, bfrag[j],
                    (short)0, acc[i][j], false, false);
    }

    // epilogue: bias + non-temporal stores (don't evict f16 operands from L2)
    float bj[4];
#pragma unroll
    for (int j = 0; j < 4; ++j)
        bj[j] = bias[nTile + wn * 64 + j * 16 + l16];

#pragma unroll
    for (int i = 0; i < 4; ++i) {
#pragma unroll
        for (int j = 0; j < 4; ++j) {
            const int ncol = nTile + wn * 64 + j * 16 + l16;
#pragma unroll
            for (int r = 0; r < 8; ++r) {
                const int mrow = mTile + wm * 64 + i * 16 + lgrp * 8 + r;
                __builtin_nontemporal_store(acc[i][j][r] + bj[j],
                                            &out[(size_t)mrow * OUT_F + ncol]);
            }
        }
    }
}

// ===========================================================================
// Fallback: fused dequant GEMM (no workspace needed). 128x128 tile, BK=32.
// ===========================================================================
__global__ __launch_bounds__(256, 2)
void nf4_linear_fused_kernel(const float* __restrict__ x,
                             const int*   __restrict__ w_idx,
                             const float* __restrict__ absmax,
                             const float* __restrict__ bias,
                             float*       __restrict__ out)
{
    __shared__ __align__(16) _Float16 Alds[128 * LDA];
    __shared__ __align__(16) _Float16 Blds[128 * LDA];
    __shared__ float codeLds[16];

    const int tid    = threadIdx.x;
    const int lane   = tid & 31;
    const int waveId = tid >> 5;
    const int wm     = waveId >> 2;
    const int wn     = waveId & 3;
    const int lgrp   = lane >> 4;
    const int l16    = lane & 15;

    const int mTile = blockIdx.y * 128;
    const int nTile = blockIdx.x * 128;

    if (tid < 16) codeLds[tid] = NF4_CODE_D[tid];
    __syncthreads();

    v4f xr[4]; v4i ir[4]; float am[4];
    const float* xBase = x     + (size_t)mTile * IN_F;
    const int*   iBase = w_idx + (size_t)nTile * IN_F;

    auto load_tile = [&](int ks) {
        const float* xp = xBase + ks * 32;
        const int*   ip = iBase + ks * 32;
#pragma unroll
        for (int r = 0; r < 4; ++r) {
            const int f = r * 256 + tid, row = f >> 3, c4 = f & 7;
            xr[r] = *(const v4f*)(xp + (size_t)row * IN_F + c4 * 4);
            ir[r] = *(const v4i*)(ip + (size_t)row * IN_F + c4 * 4);
            am[r] = absmax[(size_t)(nTile + row) * (IN_F / QBLOCK)
                           + ((ks * 32 + c4 * 4) >> 6)];
        }
    };
    auto store_tile = [&]() {
#pragma unroll
        for (int r = 0; r < 4; ++r) {
            const int f = r * 256 + tid, row = f >> 3, c4 = f & 7;
            v4h ah;
            ah.x = (_Float16)xr[r].x;  ah.y = (_Float16)xr[r].y;
            ah.z = (_Float16)xr[r].z;  ah.w = (_Float16)xr[r].w;
            *(v4h*)&Alds[row * LDA + c4 * 4] = ah;
            const float s = am[r];
            v4h bh;
            bh.x = (_Float16)(codeLds[ir[r].x] * s);
            bh.y = (_Float16)(codeLds[ir[r].y] * s);
            bh.z = (_Float16)(codeLds[ir[r].z] * s);
            bh.w = (_Float16)(codeLds[ir[r].w] * s);
            *(v4h*)&Blds[row * LDA + c4 * 4] = bh;
        }
    };

    v8f acc[4][2];
#pragma unroll
    for (int i = 0; i < 4; ++i)
#pragma unroll
        for (int j = 0; j < 2; ++j) acc[i][j] = 0.f;

    load_tile(0);
    store_tile();

    for (int ks = 0; ks < KSTEPS; ++ks) {
        __syncthreads();
        if (ks + 1 < KSTEPS) load_tile(ks + 1);

        v16h afrag[4];
#pragma unroll
        for (int i = 0; i < 4; ++i) {
            const int mrow = wm * 64 + i * 16 + l16;
            FragU ua;
            ua.h[0] = *(const v8h*)&Alds[mrow * LDA + lgrp * 8];
            ua.h[1] = *(const v8h*)&Alds[mrow * LDA + 16 + lgrp * 8];
            afrag[i] = ua.v;
        }
        v16h bfrag[2];
#pragma unroll
        for (int j = 0; j < 2; ++j) {
            const int nrow = wn * 32 + j * 16 + l16;
            FragU ub;
            ub.h[0] = *(const v8h*)&Blds[nrow * LDA + lgrp * 8];
            ub.h[1] = *(const v8h*)&Blds[nrow * LDA + 16 + lgrp * 8];
            bfrag[j] = ub.v;
        }
#pragma unroll
        for (int i = 0; i < 4; ++i)
#pragma unroll
            for (int j = 0; j < 2; ++j)
                acc[i][j] = __builtin_amdgcn_wmma_f32_16x16x32_f16(
                    false, afrag[i], false, bfrag[j],
                    (short)0, acc[i][j], false, false);

        __syncthreads();
        if (ks + 1 < KSTEPS) store_tile();
    }

    float bj[2];
#pragma unroll
    for (int j = 0; j < 2; ++j)
        bj[j] = bias[nTile + wn * 32 + j * 16 + l16];
#pragma unroll
    for (int i = 0; i < 4; ++i)
#pragma unroll
        for (int j = 0; j < 2; ++j) {
            const int ncol = nTile + wn * 32 + j * 16 + l16;
#pragma unroll
            for (int r = 0; r < 8; ++r) {
                const int mrow = mTile + wm * 64 + i * 16 + lgrp * 8 + r;
                __builtin_nontemporal_store(acc[i][j][r] + bj[j],
                                            &out[(size_t)mrow * OUT_F + ncol]);
            }
        }
}

// ===========================================================================
extern "C" void kernel_launch(void* const* d_in, const int* in_sizes, int n_in,
                              void* d_out, int out_size, void* d_ws, size_t ws_size,
                              hipStream_t stream) {
    const float* x      = (const float*)d_in[0];   // [8192, 4096] f32
    const int*   w_idx  = (const int*)d_in[1];     // [11008*4096] int32 (0..15)
    const float* absmax = (const float*)d_in[2];   // [11008*64] f32
    const float* bias   = (const float*)d_in[3];   // [11008] f32
    float*       out    = (float*)d_out;           // [8192, 11008] f32

    const size_t whElems = (size_t)OUT_F * IN_F;       // 45,088,768
    const size_t xhElems = (size_t)M_TOTAL * IN_F;     // 33,554,432
    const size_t need    = (whElems + xhElems) * sizeof(_Float16);  // ~157 MB

    if (ws_size >= need) {
        _Float16* wh = (_Float16*)d_ws;
        _Float16* xh = wh + whElems;
        nf4_dequant_w_kernel<<<dim3((unsigned)(whElems / 2048)), dim3(256), 0, stream>>>(
            w_idx, absmax, wh);
        x_to_f16_kernel<<<dim3((unsigned)(xhElems / 2048)), dim3(256), 0, stream>>>(
            x, xh);
        f16_gemm_wmma_kernel<<<dim3(OUT_F / GBN, M_TOTAL / GBM), dim3(256), 0, stream>>>(
            xh, wh, bias, out);
    } else {
        nf4_linear_fused_kernel<<<dim3(OUT_F / 128, M_TOTAL / 128), dim3(256), 0, stream>>>(
            x, w_idx, absmax, bias, out);
    }
}
